// Attention_7679401525782
// MI455X (gfx1250) — matrix operand order repaired and checked by reading, not compiled, not run on hardware
//
#include <hip/hip_runtime.h>

// ---------------------------------------------------------------------------
// MI455X (gfx1250) fused attention: QKV projection + flash attention.
// Compute-bound (~3.2e11 FLOPs vs ~300MB HBM traffic) -> everything routed
// through v_wmma_f32_16x16x32_f16 (f16 inputs, f32 accumulate).
// ---------------------------------------------------------------------------

typedef _Float16 v16h __attribute__((ext_vector_type(16)));
typedef _Float16 v8h  __attribute__((ext_vector_type(8)));
typedef float    v8f  __attribute__((ext_vector_type(8)));
typedef float    v4f  __attribute__((ext_vector_type(4)));

#define D_H 768
#define SEQ 2048
#define NBATCH 16
#define QSCALE 0.03608439182435161f   // 1/sqrt(768)

static __device__ __forceinline__ v8f wmma16(v16h a, v16h b, v8f c) {
    // D = A(16x32 f16) * B(32x16 f16) + C(16x16 f32)
    return __builtin_amdgcn_wmma_f32_16x16x32_f16(
        /*neg_a=*/false, a, /*neg_b=*/false, b,
        /*c_mod=*/(short)0, c, /*reuse_a=*/false, /*reuse_b=*/false);
}

// ---------------------------------------------------------------------------
// Kernel 1: convert f32 weight [D,H] -> f16 transposed [H,D]
// ---------------------------------------------------------------------------
__global__ void wconv_kernel(const float* __restrict__ W, _Float16* __restrict__ WT) {
    int idx = blockIdx.x * 256 + threadIdx.x;      // idx < 768*768
    int d = idx / D_H, h = idx % D_H;
    WT[h * D_H + d] = (_Float16)W[idx];
}

// ---------------------------------------------------------------------------
// Kernel 2: QKV projection GEMM. M = B*SEQ = 32768, N = K = 768.
// grid = (M/16, 3), block = 256 (8 waves). Block owns a 16-row M tile and all
// 768 N columns; wave w owns N strip [96w, 96w+96) (6 tiles). One A-fragment
// load+convert per 32-K step feeds 6 WMMAs; B rows are wave-private.
// z==0: q (pre-scaled by 1/sqrt(H)), z==1: k (row-major), z==2: v (transposed).
// ---------------------------------------------------------------------------
__global__ void __launch_bounds__(256)
qkv_kernel(const float* __restrict__ x_to, const float* __restrict__ x_from,
           const _Float16* __restrict__ wqT, const _Float16* __restrict__ wkT,
           const _Float16* __restrict__ wvT,
           const float* __restrict__ bq, const float* __restrict__ bk,
           const float* __restrict__ bv,
           _Float16* __restrict__ qf, _Float16* __restrict__ kf,
           _Float16* __restrict__ vt) {
    const int z = blockIdx.y;
    const float*    X    = (z == 0) ? x_to : x_from;
    const _Float16* WT   = (z == 0) ? wqT : ((z == 1) ? wkT : wvT);
    const float*    bias = (z == 0) ? bq  : ((z == 1) ? bk  : bv);

    const int lane = threadIdx.x & 31;
    const int wv_  = threadIdx.x >> 5;
    const int lh   = lane >> 4;           // lane half (0/1)
    const int ln   = lane & 15;
    const int m0   = blockIdx.x * 16;
    const int n0   = wv_ * 96;            // this wave's 6 N-tiles

    // A-frag row (M = lane%16); K pattern: e + lh*16
    const float* aptr = X + (size_t)(m0 + ln) * D_H + lh * 16;
    // B-frag rows: N = lane%16 fixed row of WT, 16 contiguous K halves (+lh*16)
    const _Float16* bptr[6];
#pragma unroll
    for (int j = 0; j < 6; j++)
        bptr[j] = WT + (size_t)(n0 + j * 16 + ln) * D_H + lh * 16;

    v8f acc[6];
#pragma unroll
    for (int j = 0; j < 6; j++) acc[j] = (v8f){};

    for (int kk = 0; kk < D_H; kk += 32) {
        v4f a0 = *(const v4f*)(aptr + kk);
        v4f a1 = *(const v4f*)(aptr + kk + 4);
        v4f a2 = *(const v4f*)(aptr + kk + 8);
        v4f a3 = *(const v4f*)(aptr + kk + 12);
        v16h af;
#pragma unroll
        for (int i = 0; i < 4; i++) {
            af[i]      = (_Float16)a0[i];
            af[4 + i]  = (_Float16)a1[i];
            af[8 + i]  = (_Float16)a2[i];
            af[12 + i] = (_Float16)a3[i];
        }
#pragma unroll
        for (int j = 0; j < 6; j++) {
            v16h bf = *(const v16h*)(bptr[j] + kk);   // two 16B loads
            acc[j] = wmma16(af, bf, acc[j]);
        }
    }

    const float scale = (z == 0) ? QSCALE : 1.0f;
    const int   bb    = m0 / SEQ;
    const int   qbase = (m0 % SEQ) + lh * 8;
#pragma unroll
    for (int j = 0; j < 6; j++) {
        const int n = n0 + j * 16 + ln;
        const float bval = bias[n];
#pragma unroll
        for (int r = 0; r < 8; r++) acc[j][r] = (acc[j][r] + bval) * scale;

        if (z < 2) {
            _Float16* out = (z == 0) ? qf : kf;       // row-major [M][768]
#pragma unroll
            for (int r = 0; r < 8; r++) {
                int m = m0 + r + lh * 8;              // C/D layout: M = r + lh*8
                out[(size_t)m * D_H + n] = (_Float16)acc[j][r];
            }
        } else {
            // V transposed: vt[(b*768 + n)*2048 + qpos]; 8 contiguous halves
            v8h hv;
#pragma unroll
            for (int r = 0; r < 8; r++) hv[r] = (_Float16)acc[j][r];
            *(v8h*)(vt + (size_t)(bb * D_H + n) * SEQ + qbase) = hv;
        }
    }
}

// ---------------------------------------------------------------------------
// Kernel 3: flash attention. grid = (SEQ/16, B), block = 256 (8 waves).
// Block owns a 16-row q tile; wave w owns H-chunk [96w, 96w+96).
// Per 64-k block: partial S per wave (12 WMMA) -> LDS reduce -> online
// softmax stats (wave0) -> P frags via exp -> rescale O + PV (12 WMMA).
// ---------------------------------------------------------------------------
__global__ void __launch_bounds__(256)
attn_kernel(const _Float16* __restrict__ qf, const _Float16* __restrict__ kf,
            const _Float16* __restrict__ vt, float* __restrict__ out) {
    __shared__ __align__(32) float Spart[8][1024];   // [wave][t*256 + n*16 + m]
    __shared__ __align__(32) float Sfin[1024];       // [t*256 + n*16 + m]
    __shared__ float rowM[16], rowL[16], rowA[16];

    const int tid  = threadIdx.x;
    const int lane = tid & 31;
    const int wv_  = tid >> 5;
    const int lh   = lane >> 4;
    const int ln   = lane & 15;
    const int b    = blockIdx.y;
    const int q0   = blockIdx.x * 16;
    const int h0   = wv_ * 96;

    // Preload Q A-frags for this wave's H-chunk (already scaled by 1/sqrt(H)).
    v16h qa[3];
    {
        const _Float16* qrow = qf + (size_t)(b * SEQ + q0 + ln) * D_H + lh * 16;
#pragma unroll
        for (int f = 0; f < 3; f++) {
            const _Float16* p = qrow + h0 + f * 32;
            v8h c0 = *(const v8h*)(p);
            v8h c1 = *(const v8h*)(p + 8);
#pragma unroll
            for (int i = 0; i < 8; i++) { qa[f][i] = c0[i]; qa[f][8 + i] = c1[i]; }
        }
    }

    v8f oacc[6];
#pragma unroll
    for (int j = 0; j < 6; j++) oacc[j] = (v8f){};

    if (tid < 16) { rowM[tid] = -3.0e38f; rowL[tid] = 0.0f; }
    __syncthreads();

    for (int kb = 0; kb < SEQ; kb += 64) {
        // --- 1. partial S tiles over this wave's H-chunk ---
#pragma unroll
        for (int t = 0; t < 4; t++) {
            v8f s = {};
            const _Float16* kp =
                kf + (size_t)(b * SEQ + kb + t * 16 + ln) * D_H + h0 + lh * 16;
#pragma unroll
            for (int f = 0; f < 3; f++) {
                v16h kfrag = *(const v16h*)(kp + f * 32);
                s = wmma16(qa[f], kfrag, s);
            }
            *(v8f*)&Spart[wv_][t * 256 + ln * 16 + lh * 8] = s;  // 2x ds_store_b128
        }

        // Prefetch next k-block's K/V rows (global_prefetch_b8) to overlap the
        // LDS-reduction + softmax phase, during which no global loads issue.
        if (kb + 64 < SEQ) {
            __builtin_prefetch(
                kf + (size_t)(b * SEQ + kb + 64 + ln) * D_H + h0 + lh * 16, 0, 3);
            __builtin_prefetch(
                vt + (size_t)(b * D_H + h0 + ln) * SEQ + kb + 64 + lh * 16, 0, 3);
        }
        __syncthreads();   // also protects prev-iter Sfin/rowM consumers

        // --- 2. reduce 8 partial S blocks -> Sfin ---
#pragma unroll
        for (int i = 0; i < 4; i++) {
            int idx = tid + i * 256;
            float s = 0.0f;
#pragma unroll
            for (int w2 = 0; w2 < 8; w2++) s += Spart[w2][idx];
            Sfin[idx] = s;
        }
        __syncthreads();

        // --- 3. online softmax row stats (wave0, one lane per q row) ---
        if (tid < 16) {
            float mold = rowM[tid];
            float mb = mold;
            for (int c = 0; c < 64; c++)
                mb = fmaxf(mb, Sfin[(c >> 4) * 256 + (c & 15) * 16 + tid]);
            float alpha = __expf(mold - mb);
            float lsum = 0.0f;
            for (int c = 0; c < 64; c++)
                lsum += __expf(Sfin[(c >> 4) * 256 + (c & 15) * 16 + tid] - mb);
            rowL[tid] = rowL[tid] * alpha + lsum;
            rowM[tid] = mb;
            rowA[tid] = alpha;
        }
        __syncthreads();

        // --- 4. build P A-frags (redundant per wave; 32 exps/lane) ---
        v16h pf[2];
        {
            float mrow = rowM[ln];      // A layout: q row = lane%16
#pragma unroll
            for (int t2 = 0; t2 < 2; t2++) {
#pragma unroll
                for (int e = 0; e < 16; e++) {
                    int kl = t2 * 32 + e + lh * 16;
                    float s = Sfin[(kl >> 4) * 256 + (kl & 15) * 16 + ln];
                    pf[t2][e] = (_Float16)__expf(s - mrow);
                }
            }
        }
        float alr[8];
#pragma unroll
        for (int r = 0; r < 8; r++) alr[r] = rowA[r + lh * 8];

        // --- 5. rescale O and accumulate P*V for this wave's H-chunk ---
#pragma unroll
        for (int j = 0; j < 6; j++) {
#pragma unroll
            for (int r = 0; r < 8; r++) oacc[j][r] *= alr[r];
            const _Float16* vp =
                vt + (size_t)(b * D_H + h0 + j * 16 + ln) * SEQ + kb + lh * 16;
#pragma unroll
            for (int t2 = 0; t2 < 2; t2++) {
                v16h vfrag = *(const v16h*)(vp + t2 * 32);
                oacc[j] = wmma16(pf[t2], vfrag, oacc[j]);
            }
        }
    }

    // --- epilogue: normalize by row sums, store f32 output ---
    float linv[8];
#pragma unroll
    for (int r = 0; r < 8; r++) linv[r] = 1.0f / rowL[r + lh * 8];
#pragma unroll
    for (int j = 0; j < 6; j++) {
        int hcol = h0 + j * 16 + ln;
#pragma unroll
        for (int r = 0; r < 8; r++) {
            int qrow = q0 + r + lh * 8;
            out[(size_t)(b * SEQ + qrow) * D_H + hcol] = oacc[j][r] * linv[r];
        }
    }
}

// ---------------------------------------------------------------------------
extern "C" void kernel_launch(void* const* d_in, const int* in_sizes, int n_in,
                              void* d_out, int out_size, void* d_ws, size_t ws_size,
                              hipStream_t stream) {
    const float* x_to   = (const float*)d_in[0];
    const float* x_from = (const float*)d_in[1];
    const float* Wq     = (const float*)d_in[2];
    const float* bq     = (const float*)d_in[3];
    const float* Wk     = (const float*)d_in[4];
    const float* bk     = (const float*)d_in[5];
    const float* Wv     = (const float*)d_in[6];
    const float* bv     = (const float*)d_in[7];
    float* out = (float*)d_out;

    // Workspace layout (f16 elements): 3 transposed weights + q + k + v^T
    _Float16* ws = (_Float16*)d_ws;
    const size_t WSZ = (size_t)D_H * D_H;          // 589824
    const size_t QSZ = (size_t)NBATCH * SEQ * D_H; // 25165824
    _Float16* wqT = ws;
    _Float16* wkT = wqT + WSZ;
    _Float16* wvT = wkT + WSZ;
    _Float16* qf  = wvT + WSZ;
    _Float16* kf  = qf + QSZ;
    _Float16* vt  = kf + QSZ;                      // total ~147.5 MB

    wconv_kernel<<<(D_H * D_H) / 256, 256, 0, stream>>>(Wq, wqT);
    wconv_kernel<<<(D_H * D_H) / 256, 256, 0, stream>>>(Wk, wkT);
    wconv_kernel<<<(D_H * D_H) / 256, 256, 0, stream>>>(Wv, wvT);

    // M/16 = 32768/16 = 2048 tiles; each block covers all 768 N; z picks q/k/v
    qkv_kernel<<<dim3(2048, 3), 256, 0, stream>>>(
        x_to, x_from, wqT, wkT, wvT, bq, bk, bv, qf, kf, vt);

    attn_kernel<<<dim3(SEQ / 16, NBATCH), 256, 0, stream>>>(qf, kf, vt, out);
}
